// P3_DeeperGCN_19112604467650
// MI455X (gfx1250) — matrix operand order repaired and checked by reading, not compiled, not run on hardware
//
#include <hip/hip_runtime.h>

// ---------------------------------------------------------------------------
// DeeperGCN forward for MI455X (gfx1250, wave32, WMMA + TDM).
//
//   1. CSR build (histogram -> scan -> scatter); single-pass online-softmax
//      aggregation (no float atomics; h is L2-resident: 51 MB << 192 MB L2).
//   2. All GEMMs in f16 WMMA (v_wmma_f32_16x16x32_f16) with f32 accum;
//      A/B tiles staged into LDS by the Tensor Data Mover (TENSOR_LOAD_TO_LDS,
//      TENSORcnt), with descriptor padding giving a conflict-free 136-elem
//      LDS pitch for the fragment ds_load_b128s.
//   3. LN / aggregation / log-softmax use one wave per node with shuffle
//      reductions (no LDS, no block barriers).
// ---------------------------------------------------------------------------

typedef __attribute__((ext_vector_type(16))) _Float16 v16h;
typedef __attribute__((ext_vector_type(8)))  _Float16 v8h;
typedef __attribute__((ext_vector_type(4)))  _Float16 v4h;
typedef __attribute__((ext_vector_type(8)))  float    v8f;
typedef __attribute__((ext_vector_type(4)))  float    v4f;
typedef __attribute__((ext_vector_type(4)))  unsigned u32x4;
typedef __attribute__((ext_vector_type(4)))  int      i32x4;
typedef __attribute__((ext_vector_type(8)))  int      i32x8;

#define HDIM    128
#define LDT     136   // LDS pitch in f16: 128 data + 8 pad (272B rows, 16B aligned, 4-bank stride)
#define MSG_EPS 1e-7f
#define LN_EPS  1e-5f

#define HAVE_TDM __has_builtin(__builtin_amdgcn_tensor_load_to_lds)

// ------------------------------ CSR build ----------------------------------

__global__ void count_kernel(const int* __restrict__ row, int* __restrict__ deg, int E) {
    int e = blockIdx.x * 256 + threadIdx.x;
    if (e < E) atomicAdd(&deg[row[e]], 1);
}

__global__ __launch_bounds__(1024) void scan_kernel(const int* __restrict__ deg,
                                                    int* __restrict__ rowptr, int n) {
    __shared__ int buf[1024];
    const int tid = threadIdx.x;
    int carry = 0;
    if (tid == 0) rowptr[0] = 0;
    for (int base = 0; base < n; base += 1024) {
        int i = base + tid;
        buf[tid] = (i < n) ? deg[i] : 0;
        __syncthreads();
        for (int off = 1; off < 1024; off <<= 1) {
            int t = (tid >= off) ? buf[tid - off] : 0;
            __syncthreads();
            buf[tid] += t;
            __syncthreads();
        }
        if (i < n) rowptr[i + 1] = carry + buf[tid];
        carry += buf[1023];
        __syncthreads();
    }
}

__global__ void scatter_kernel(const int* __restrict__ row, const int* __restrict__ col,
                               int* __restrict__ cursor, int* __restrict__ colidx, int E) {
    int e = blockIdx.x * 256 + threadIdx.x;
    if (e < E) {
        int p = atomicAdd(&cursor[row[e]], 1);
        colidx[p] = col[e];
    }
}

// ------------------------- f16 conversion helpers --------------------------

__global__ void cvt_f16_kernel(const float* __restrict__ src, _Float16* __restrict__ dst, int n8) {
    int i = blockIdx.x * 256 + threadIdx.x;   // one v8h per thread
    if (i < n8) {
        int idx = i * 8;
        v4f a = *(const v4f*)&src[idx];
        v4f b = *(const v4f*)&src[idx + 4];
        v8h o;
        o[0] = (_Float16)a[0]; o[1] = (_Float16)a[1]; o[2] = (_Float16)a[2]; o[3] = (_Float16)a[3];
        o[4] = (_Float16)b[0]; o[5] = (_Float16)b[1]; o[6] = (_Float16)b[2]; o[7] = (_Float16)b[3];
        *(v8h*)&dst[idx] = o;
    }
}

// BT[n][k] = (f16) W[k][n], zero-padded to nt16 rows.
__global__ void prep_wt_kernel(const float* __restrict__ W, _Float16* __restrict__ BT,
                               int K, int bcols, int nt16) {
    int idx = blockIdx.x * 256 + threadIdx.x;
    if (idx < nt16 * K) {
        int nrow = idx / K;
        int k = idx - nrow * K;
        BT[idx] = (nrow < bcols) ? (_Float16)W[(size_t)k * bcols + nrow] : (_Float16)0.f;
    }
}

// --------------------- GENConv softmax aggregation -------------------------
// One wave per node, 4 features per lane, online softmax over the neighbor
// list (single gather pass). Writes the GEMM input directly as f16.
__global__ __launch_bounds__(256) void agg_kernel(const float* __restrict__ hx,
                                                  const int* __restrict__ rowptr,
                                                  const int* __restrict__ colidx,
                                                  const float* __restrict__ t, int layer,
                                                  _Float16* __restrict__ gih, int n) {
    const int wave = threadIdx.x >> 5;
    const int lane = threadIdx.x & 31;
    const int v = blockIdx.x * 8 + wave;
    if (v >= n) return;                     // wave-uniform
    const float tv = t[layer];
    const int s0 = rowptr[v];
    const int s1 = rowptr[v + 1];
    const int fb = lane * 4;

    float M[4], D[4], Nm[4];
    #pragma unroll
    for (int c = 0; c < 4; ++c) { M[c] = -__builtin_inff(); D[c] = 0.f; Nm[c] = 0.f; }

    for (int e = s0; e < s1; ++e) {
        int j = colidx[e];                  // uniform -> scalar load
        v4f hv = *(const v4f*)&hx[(size_t)j * HDIM + fb];
        #pragma unroll
        for (int c = 0; c < 4; ++c) {
            float m = fmaxf(hv[c], 0.f) + MSG_EPS;
            float s = m * tv;
            if (s > M[c]) {                 // rescale (exp(-inf)=0 covers first edge)
                float r = __expf(M[c] - s);
                D[c] *= r; Nm[c] *= r; M[c] = s;
            }
            float ex = __expf(s - M[c]);
            D[c] += ex;
            Nm[c] += ex * m;
        }
    }
    v4f self = *(const v4f*)&hx[(size_t)v * HDIM + fb];
    v4h o;
    #pragma unroll
    for (int c = 0; c < 4; ++c)
        o[c] = (_Float16)(self[c] + Nm[c] / fmaxf(D[c], 1e-16f));
    *(v4h*)&gih[(size_t)v * HDIM + fb] = o;
}

// --------------------------- LayerNorm + ReLU ------------------------------
// One wave per node; shuffle reductions. Emits f32 (for the edge gather) and
// f16 (for WMMA) copies.
__global__ __launch_bounds__(256) void ln_relu_kernel(const float* __restrict__ h,
                                                      const float* __restrict__ gamma,
                                                      const float* __restrict__ beta,
                                                      float* __restrict__ hx,
                                                      _Float16* __restrict__ hxh, int n) {
    const int wave = threadIdx.x >> 5;
    const int lane = threadIdx.x & 31;
    const int v = blockIdx.x * 8 + wave;
    if (v >= n) return;
    const int fb = lane * 4;
    v4f x = *(const v4f*)&h[(size_t)v * HDIM + fb];
    float s = x[0] + x[1] + x[2] + x[3];
    #pragma unroll
    for (int off = 16; off > 0; off >>= 1) s += __shfl_xor(s, off, 32);
    float mu = s * (1.f / 128.f);
    v4f d;
    float sq = 0.f;
    #pragma unroll
    for (int c = 0; c < 4; ++c) { d[c] = x[c] - mu; sq += d[c] * d[c]; }
    #pragma unroll
    for (int off = 16; off > 0; off >>= 1) sq += __shfl_xor(sq, off, 32);
    float inv = rsqrtf(sq * (1.f / 128.f) + LN_EPS);
    v4f g = *(const v4f*)&gamma[fb];
    v4f b = *(const v4f*)&beta[fb];
    v4f y;
    v4h yh;
    #pragma unroll
    for (int c = 0; c < 4; ++c) {
        float t0 = fmaxf(d[c] * inv * g[c] + b[c], 0.f);
        y[c] = t0;
        yh[c] = (_Float16)t0;
    }
    *(v4f*)&hx[(size_t)v * HDIM + fb] = y;
    *(v4h*)&hxh[(size_t)v * HDIM + fb] = yh;
}

// ----------------------- TDM tile load (2D, padded) ------------------------

#if HAVE_TDM
__device__ __forceinline__ void tdm_load_2d(unsigned lds_off, const _Float16* gsrc,
                                            unsigned rows_tile, unsigned rows_avail,
                                            unsigned row_stride_elems) {
    unsigned long long ga = (unsigned long long)(size_t)gsrc;
    u32x4 g0;
    g0[0] = 1u;                                                // count=1 (valid), user mode
    g0[1] = lds_off;                                           // LDS destination byte address
    g0[2] = (unsigned)ga;                                      // global_addr[31:0]
    g0[3] = (unsigned)((ga >> 32) & 0x01ffffffu) | (2u << 30); // global_addr[56:32] | type=2
    // data_size=1 (2B); pad_enable; pad_interval code 5 (=64 dwords = 128 f16);
    // pad_amount code 3 (=4 dwords = 8 f16)  ->  LDS pitch 136 f16 = LDT.
    i32x8 g1;
    unsigned td0 = 128u;                                       // chunk treated as its own tensor
    unsigned td1 = rows_avail;                                 // OOB rows read as zero
    g1[0] = (int)((1u << 16) | (1u << 20) | (5u << 22) | (3u << 25));
    g1[1] = (int)((td0 & 0xffffu) << 16);                      // tensor_dim0[15:0]
    g1[2] = (int)((td0 >> 16) | ((td1 & 0xffffu) << 16));      // td0[31:16] | td1[15:0]
    g1[3] = (int)((td1 >> 16) | (128u << 16));                 // td1[31:16] | tile_dim0=128
    g1[4] = (int)(rows_tile & 0xffffu);                        // tile_dim1 | tile_dim2=0
    g1[5] = (int)row_stride_elems;                             // tensor_dim0_stride[31:0]
    g1[6] = 0;                                                 // stride0[47:32] | stride1 lo
    g1[7] = 0;
    i32x4 z4; z4[0] = z4[1] = z4[2] = z4[3] = 0;
#if __clang_major__ >= 23
    i32x8 z8;
    #pragma unroll
    for (int i = 0; i < 8; ++i) z8[i] = 0;
    __builtin_amdgcn_tensor_load_to_lds(g0, g1, z4, z4, z8, 0);
#else
    __builtin_amdgcn_tensor_load_to_lds(g0, g1, z4, z4, 0);
#endif
}
#endif

// ------------------------------- WMMA GEMM ---------------------------------
// out[nrows, ldout] = A[nrows, KTOT] (f16) @ BT^T + bias (+ resid)
// BT is the pre-transposed f16 weight tensor [NT*16, KTOT].
// 256 threads = 8 waves: 4 row-blocks of 16 x 2 column halves.
template <int KTOT, int NT, bool RESID>
__global__ __launch_bounds__(256) void gemm_wmma_kernel(const _Float16* __restrict__ A,
                                                        const _Float16* __restrict__ BT,
                                                        const float* __restrict__ bias,
                                                        const float* __restrict__ resid,
                                                        float* __restrict__ out,
                                                        int nrows, int bcols, int ldout) {
    __shared__ __align__(16) _Float16 Alds[64 * LDT];
    __shared__ __align__(16) _Float16 Blds[NT * 16 * LDT];

    const int tid     = threadIdx.x;
    const int wave    = tid >> 5;
    const int lane    = tid & 31;
    const int hf      = lane >> 4;
    const int ln      = lane & 15;
    const int rowblk  = wave >> 1;
    const int colhalf = wave & 1;
    const int rowbase = blockIdx.x * 64;
    constexpr int NTW = (NT + 1) / 2;

    v8f acc[NTW];
    #pragma unroll
    for (int c = 0; c < NTW; ++c)
        #pragma unroll
        for (int i = 0; i < 8; ++i) acc[c][i] = 0.f;

    for (int kc = 0; kc < KTOT / 128; ++kc) {
        __syncthreads();                        // previous chunk's compute done
#if HAVE_TDM
        if (wave == 0) {
            tdm_load_2d((unsigned)(size_t)&Alds[0],
                        A + (size_t)rowbase * KTOT + kc * 128,
                        64u, (unsigned)(nrows - rowbase), (unsigned)KTOT);
            tdm_load_2d((unsigned)(size_t)&Blds[0],
                        BT + kc * 128,
                        (unsigned)(NT * 16), (unsigned)(NT * 16), (unsigned)KTOT);
            __builtin_amdgcn_s_wait_tensorcnt(0);
        }
#else
        // fallback: plain f16 copies through VGPRs
        for (int idx = tid; idx < 64 * 16; idx += 256) {       // 16 v8h per row
            int r = idx >> 4, kk = (idx & 15) << 3;
            int rg = rowbase + r;
            v8h val;
            #pragma unroll
            for (int i = 0; i < 8; ++i) val[i] = (_Float16)0.f;
            if (rg < nrows) val = *(const v8h*)&A[(size_t)rg * KTOT + kc * 128 + kk];
            *(v8h*)&Alds[r * LDT + kk] = val;
        }
        for (int idx = tid; idx < NT * 16 * 16; idx += 256) {
            int r = idx >> 4, kk = (idx & 15) << 3;
            *(v8h*)&Blds[r * LDT + kk] =
                *(const v8h*)&BT[(size_t)r * KTOT + kc * 128 + kk];
        }
#endif
        __syncthreads();

        #pragma unroll
        for (int kt = 0; kt < 4; ++kt) {
            // A fragment: lane holds row ln, K = kt*32 + hf*8 + {0..7, 16..23}
            const _Float16* ap = &Alds[(rowblk * 16 + ln) * LDT + kt * 32 + hf * 8];
            v8h alo = *(const v8h*)ap;
            v8h ahi = *(const v8h*)(ap + 16);
            v16h af;
            #pragma unroll
            for (int i = 0; i < 8; ++i) { af[i] = alo[i]; af[i + 8] = ahi[i]; }

            #pragma unroll
            for (int j = 0; j < NTW; ++j) {
                int c = colhalf * NTW + j;      // wave-uniform
                if (c < NT) {
                    const _Float16* bp = &Blds[(c * 16 + ln) * LDT + kt * 32 + hf * 8];
                    v8h blo = *(const v8h*)bp;
                    v8h bhi = *(const v8h*)(bp + 16);
                    v16h bf;
                    #pragma unroll
                    for (int i = 0; i < 8; ++i) { bf[i] = blo[i]; bf[i + 8] = bhi[i]; }
                    acc[j] = __builtin_amdgcn_wmma_f32_16x16x32_f16(
                        false, af, false, bf, (short)0, acc[j], false, false);
                }
            }
        }
    }

    // store: D VGPR i -> row hf*8+i, col ln
    #pragma unroll
    for (int j = 0; j < NTW; ++j) {
        int c = colhalf * NTW + j;
        if (c >= NT) continue;
        int colg = c * 16 + ln;
        float bv = (colg < bcols) ? bias[colg] : 0.f;
        #pragma unroll
        for (int i = 0; i < 8; ++i) {
            int rg = rowbase + rowblk * 16 + hf * 8 + i;
            if (rg < nrows && colg < bcols) {
                float v = acc[j][i] + bv;
                if constexpr (RESID) v += resid[(size_t)rg * HDIM + colg];
                out[(size_t)rg * ldout + colg] = v;
            }
        }
    }
}

// ------------------------------ log-softmax --------------------------------

__global__ __launch_bounds__(256) void logsoftmax_kernel(float* __restrict__ out, int n) {
    const int wave = threadIdx.x >> 5;
    const int lane = threadIdx.x & 31;
    const int v = blockIdx.x * 8 + wave;
    if (v >= n) return;
    float v0 = (lane < 47) ? out[(size_t)v * 47 + lane] : -__builtin_inff();
    float v1 = (lane + 32 < 47) ? out[(size_t)v * 47 + lane + 32] : -__builtin_inff();
    float m = fmaxf(v0, v1);
    #pragma unroll
    for (int off = 16; off > 0; off >>= 1) m = fmaxf(m, __shfl_xor(m, off, 32));
    float e = ((lane < 47) ? __expf(v0 - m) : 0.f) + ((lane + 32 < 47) ? __expf(v1 - m) : 0.f);
    #pragma unroll
    for (int off = 16; off > 0; off >>= 1) e += __shfl_xor(e, off, 32);
    float lse = __logf(e);
    if (lane < 47)      out[(size_t)v * 47 + lane]      = v0 - m - lse;
    if (lane + 32 < 47) out[(size_t)v * 47 + lane + 32] = v1 - m - lse;
}

// ------------------------------- launcher ----------------------------------

extern "C" void kernel_launch(void* const* d_in, const int* in_sizes, int n_in,
                              void* d_out, int out_size, void* d_ws, size_t ws_size,
                              hipStream_t stream) {
    const float* x      = (const float*)d_in[0];
    const int*   row    = (const int*)  d_in[1];
    const int*   col    = (const int*)  d_in[2];
    const float* W_enc  = (const float*)d_in[3];
    const float* b_enc  = (const float*)d_in[4];
    const float* W_gcn  = (const float*)d_in[5];
    const float* b_gcn  = (const float*)d_in[6];
    const float* t      = (const float*)d_in[7];
    const float* gamma  = (const float*)d_in[8];
    const float* beta   = (const float*)d_in[9];
    const float* W_pred = (const float*)d_in[10];
    const float* b_pred = (const float*)d_in[11];
    float* out = (float*)d_out;

    const int N = in_sizes[0] / 512;
    const int E = in_sizes[1];

    char* p = (char*)d_ws;
    auto take = [&](size_t bytes) -> char* {
        char* q = p;
        p += (bytes + 255) & ~(size_t)255;
        return q;
    };
    float*     h      = (float*)    take((size_t)N * HDIM * 4);
    float*     hx     = (float*)    take((size_t)N * HDIM * 4);
    _Float16*  gih    = (_Float16*) take((size_t)N * HDIM * 2);
    _Float16*  hxh    = (_Float16*) take((size_t)N * HDIM * 2);
    _Float16*  xh     = (_Float16*) take((size_t)N * 512 * 2);
    _Float16*  BTenc  = (_Float16*) take((size_t)128 * 512 * 2);
    _Float16*  BTg    = (_Float16*) take((size_t)4 * 128 * 128 * 2);
    _Float16*  BTp    = (_Float16*) take((size_t)48 * 128 * 2);
    int*       deg    = (int*)      take((size_t)N * 4);
    int*       rowptr = (int*)      take((size_t)(N + 1) * 4);
    int*       cursor = (int*)      take((size_t)N * 4);
    int*       colidx = (int*)      take((size_t)E * 4);

    // 1) CSR build
    (void)hipMemsetAsync(deg, 0, (size_t)N * 4, stream);
    count_kernel<<<(E + 255) / 256, 256, 0, stream>>>(row, deg, E);
    scan_kernel<<<1, 1024, 0, stream>>>(deg, rowptr, N);
    (void)hipMemcpyAsync(cursor, rowptr, (size_t)N * 4, hipMemcpyDeviceToDevice, stream);
    scatter_kernel<<<(E + 255) / 256, 256, 0, stream>>>(row, col, cursor, colidx, E);

    // 2) f16 operand prep
    const int n8 = N * 512 / 8;
    cvt_f16_kernel<<<(n8 + 255) / 256, 256, 0, stream>>>(x, xh, n8);
    prep_wt_kernel<<<(128 * 512 + 255) / 256, 256, 0, stream>>>(W_enc, BTenc, 512, 128, 128);
    for (int l = 0; l < 4; ++l)
        prep_wt_kernel<<<(128 * 128 + 255) / 256, 256, 0, stream>>>(
            W_gcn + (size_t)l * 128 * 128, BTg + (size_t)l * 128 * 128, 128, 128, 128);
    prep_wt_kernel<<<(48 * 128 + 255) / 256, 256, 0, stream>>>(W_pred, BTp, 128, 47, 48);

    const int gblk = (N + 63) / 64;
    const int nblk = (N + 7) / 8;

    // 3) encoder: h = x @ W_enc + b_enc
    gemm_wmma_kernel<512, 8, false><<<gblk, 256, 0, stream>>>(
        xh, BTenc, b_enc, nullptr, h, N, HDIM, HDIM);

    // 4) layer 0 (no pre-norm, no residual)
    agg_kernel<<<nblk, 256, 0, stream>>>(h, rowptr, colidx, t, 0, gih, N);
    gemm_wmma_kernel<128, 8, false><<<gblk, 256, 0, stream>>>(
        gih, BTg, b_gcn, nullptr, h, N, HDIM, HDIM);

    // 5) layers 1..3: res+ blocks
    for (int l = 1; l < 4; ++l) {
        ln_relu_kernel<<<nblk, 256, 0, stream>>>(h, gamma + (l - 1) * HDIM,
                                                 beta + (l - 1) * HDIM, hx, hxh, N);
        agg_kernel<<<nblk, 256, 0, stream>>>(hx, rowptr, colidx, t, l, gih, N);
        gemm_wmma_kernel<128, 8, true><<<gblk, 256, 0, stream>>>(
            gih, BTg + (size_t)l * 128 * 128, b_gcn + l * HDIM, h, h, N, HDIM, HDIM);
    }

    // 6) head: LN -> ReLU -> pred GEMM -> log_softmax (in place on d_out)
    ln_relu_kernel<<<nblk, 256, 0, stream>>>(h, gamma + 3 * HDIM, beta + 3 * HDIM, hx, hxh, N);
    gemm_wmma_kernel<128, 3, false><<<gblk, 256, 0, stream>>>(
        hxh, BTp, b_pred, nullptr, out, N, 47, 47);
    logsoftmax_kernel<<<nblk, 256, 0, stream>>>(out, N);
}
